// VPINN_61366492725462
// MI455X (gfx1250) — matrix operand order repaired and checked by reading, not compiled, not run on hardware
//
#include <hip/hip_runtime.h>
#include <hip/hip_bf16.h>

typedef __attribute__((ext_vector_type(16))) _Float16 v16h;
typedef __attribute__((ext_vector_type(8)))  float    v8f;

#define E_  2048
#define S_  256
#define Q_  25
#define NN_ 36
#define R_  15
#define H_  128

// workspace layout (bytes)
#define WS_EV_OFF  0u                         // float  [E*NN]            = 294912 B
#define WS_WP_OFF  294912u                    // f16    [2][4][8][32][16] = 65536 B
#define WS_PP_OFF  360448u                    // f16    [S][3][2][32][16] = 1572864 B
#define WS_AP_OFF  1933312u                   // f16    [E/16][3][32][16] = 393216 B
#define WS_WB_OFF  2326528u                   // f16    [32][16]          = 1024 B

// ---------------------------------------------------------------------------
// tanh: native v_tanh_f32 if the toolchain exposes it, else branch-free
// sign(x) * (1 - 2/(2^(2*log2e*|x|) + 1)) via v_exp_f32 + v_rcp_f32.
// ---------------------------------------------------------------------------
__device__ __forceinline__ float fast_tanh(float x) {
#if __has_builtin(__builtin_amdgcn_tanhf)
    return __builtin_amdgcn_tanhf(x);
#else
    float a = __builtin_fabsf(x);
    float e = __builtin_amdgcn_exp2f(a * 2.88539008177792681472f); // exp(2a)
    float t = __builtin_fmaf(-2.0f, __builtin_amdgcn_rcpf(e + 1.0f), 1.0f);
    return __builtin_copysignf(t, x);
#endif
}

// ---------------------------------------------------------------------------
// Kernel 0: zero the scalar loss accumulator (deterministic per call).
// ---------------------------------------------------------------------------
__global__ void vpinn_zero_out(float* __restrict__ out, int n) {
    int i = blockIdx.x * blockDim.x + threadIdx.x;
    if (i < n) out[i] = 0.0f;
}

// ---------------------------------------------------------------------------
// WMMA fragment layouts (ISA 7.12.2, 16-bit, wave32):
//  A 16x32: lane L = row M=L%16; elem i -> v=i/2; k=(v/4)*16+(L/16)*8+(v%4)*2+(i%2)
//  B 32x16: lane L = col N=L%16; elem i -> k=(i/8)*16+(L/16)*8+(i%8)
//  C 16x16 f32: lane L col N=L%16; vgpr j -> row M=j+8*(L/16)
// ---------------------------------------------------------------------------

// Kernel P1: pack W2,W3 (f32 row-major 128x128) into B-fragment-major f16.
__global__ void vpinn_pack_w(const float* __restrict__ W2,
                             const float* __restrict__ W3,
                             _Float16* __restrict__ Wp) {
    int t = blockIdx.x * blockDim.x + threadIdx.x;
    if (t >= 2 * 32 * 32 * 16) return;
    int i = t & 15, L = (t >> 4) & 31, f = (t >> 9) & 31, l = t >> 14;
    int kt = f >> 3, nt = f & 7;
    int k = kt * 32 + ((i >> 3) << 4) + ((L >> 4) << 3) + (i & 7);
    int n = nt * 16 + (L & 15);
    const float* W = l ? W3 : W2;
    Wp[t] = (_Float16)W[k * H_ + n];
}

// Kernel P2: pack P[96,32] per sub-element s into B-fragment-major f16.
// P[k,q] = Ixx[s,q,k] (k<36) | Iyy[s,q,k-36] (k<72) | 0 ; q>=25 -> 0
__global__ void vpinn_pack_p(const float* __restrict__ Ixx,
                             const float* __restrict__ Iyy,
                             _Float16* __restrict__ Pp) {
    int t = blockIdx.x * blockDim.x + threadIdx.x;
    if (t >= S_ * 3 * 2 * 32 * 16) return;
    int i = t & 15, L = (t >> 4) & 31, tile = (t >> 9) & 1;
    int r = t >> 10;
    int kt = r % 3, s = r / 3;
    int k = kt * 32 + ((i >> 3) << 4) + ((L >> 4) << 3) + (i & 7);
    int q = tile * 16 + (L & 15);
    float v = 0.0f;
    if (q < Q_) {
        if (k < NN_)          v = Ixx[((size_t)s * Q_ + q) * NN_ + k];
        else if (k < 2 * NN_) v = Iyy[((size_t)s * Q_ + q) * NN_ + (k - NN_)];
    }
    Pp[t] = (_Float16)v;
}

// Kernel P3 (after MLP): pack B_DD-scaled ev into A-fragment-major f16.
__global__ void vpinn_pack_a(const float* __restrict__ ev,
                             const float* __restrict__ BDD,
                             _Float16* __restrict__ Ap) {
    int t = blockIdx.x * blockDim.x + threadIdx.x;
    if (t >= (E_ / 16) * 3 * 32 * 16) return;
    int i = t & 15, L = (t >> 4) & 31;
    int r = t >> 9;
    int kt = r % 3, et = r / 3;
    int v = i >> 1;
    int k = kt * 32 + ((v >> 2) << 4) + ((L >> 4) << 3) + ((v & 3) << 1) + (i & 1);
    int e = et * 16 + (L & 15);
    float val = 0.0f;
    if (k < NN_)
        val = (BDD[e * 4 + 0] + BDD[e * 4 + 2]) * ev[(size_t)e * NN_ + k];
    else if (k < 2 * NN_)
        val = (BDD[e * 4 + 1] + BDD[e * 4 + 3]) * ev[(size_t)e * NN_ + (k - NN_)];
    Ap[t] = (_Float16)val;
}

// Kernel P4: pack WB[q,r] = w[q]*Base[q,r] into a single 32x16 B-fragment f16.
__global__ void vpinn_pack_wb(const float* __restrict__ wq,
                              const float* __restrict__ Base,
                              _Float16* __restrict__ WBf) {
    int t = blockIdx.x * blockDim.x + threadIdx.x;
    if (t >= 32 * 16) return;
    int i = t & 15, L = t >> 4;
    int rr = L & 15;
    int q  = ((i >> 3) << 4) + ((L >> 4) << 3) + (i & 7);
    float v = (q < Q_ && rr < R_) ? wq[q] * Base[q * R_ + rr] : 0.0f;
    WBf[t] = (_Float16)v;
}

// ---------------------------------------------------------------------------
// Kernel 1: MLP over 73728 points (one wave32 per 16-point tile).
// ---------------------------------------------------------------------------
#define MLP_LAYER(SRC, DST, WPBASE, BB)                                         \
    {                                                                           \
        v16h af[4];                                                             \
        _Pragma("unroll") for (int kt = 0; kt < 4; ++kt) {                      \
            _Pragma("unroll") for (int i = 0; i < 16; ++i) {                    \
                int v = i >> 1;                                                 \
                int k = kt * 32 + ((v >> 2) << 4) + khalf8 + ((v & 3) << 1) + (i & 1); \
                af[kt][i] = SRC[(lane & 15) * H_ + k];                          \
            }                                                                   \
        }                                                                       \
        for (int nt = 0; nt < 8; ++nt) {                                        \
            v8f acc = {};                                                       \
            _Pragma("unroll") for (int kt = 0; kt < 4; ++kt) {                  \
                v16h b = *(const v16h*)((WPBASE) + ((kt * 8 + nt) * 32 + lane) * 16); \
                acc = __builtin_amdgcn_wmma_f32_16x16x32_f16(false, af[kt], false, b, \
                                                             (short)0, acc, false, false); \
            }                                                                   \
            int n  = nt * 16 + (lane & 15);                                     \
            float bn = BB[n];                                                   \
            _Pragma("unroll") for (int j = 0; j < 8; ++j)                       \
                DST[(mb8 + j) * H_ + n] = (_Float16)fast_tanh(acc[j] + bn);     \
        }                                                                       \
    }

__global__ __launch_bounds__(32)
void vpinn_mlp_kernel(const float* __restrict__ nodes,
                      const float* __restrict__ W1, const float* __restrict__ b1,
                      const _Float16* __restrict__ Wp,
                      const float* __restrict__ b2, const float* __restrict__ b3,
                      const float* __restrict__ W4, const float* __restrict__ b4,
                      float* __restrict__ ev) {
    __shared__ _Float16 actA[16 * H_];
    __shared__ _Float16 actB[16 * H_];
    __shared__ float    part[32];

    const int lane   = threadIdx.x;
    const int gp0    = blockIdx.x * 16;
    const int khalf8 = (lane >> 4) << 3;
    const int mb8    = (lane >> 4) << 3;

    // Layer 1 (K=2) in VALU: 2 lanes per point, 64 hidden units each
    const int p  = lane & 15;
    const int jh = (lane >> 4) * 64;
    const float x0 = nodes[(size_t)(gp0 + p) * 2 + 0];
    const float x1 = nodes[(size_t)(gp0 + p) * 2 + 1];
#pragma unroll 8
    for (int jj = 0; jj < 64; ++jj) {
        int j = jh + jj;
        float h = fast_tanh(__builtin_fmaf(x0, W1[j], __builtin_fmaf(x1, W1[H_ + j], b1[j])));
        actA[p * H_ + j] = (_Float16)h;
    }

    // Layers 2 & 3: 16x128 @ 128x128 via WMMA (pre-packed f16 weights)
    MLP_LAYER(actA, actB, Wp,         b2)
    MLP_LAYER(actB, actA, Wp + 16384, b3)

    // Layer 4 (N=1) dot product in VALU
    float s = 0.0f;
#pragma unroll 8
    for (int jj = 0; jj < 64; ++jj) {
        int j = jh + jj;
        s = __builtin_fmaf((float)actA[p * H_ + j], W4[j], s);
    }
    part[lane] = s;
    __syncthreads();
    if (lane < 16) ev[gp0 + lane] = part[lane] + part[lane + 16] + b4[0];
}

// ---------------------------------------------------------------------------
// Kernel 2: interpolation GEMM + WMMA projection + squared-residual loss.
// Grid (E/16, S), one wave32 per block.
//   integ[16e,32q] = A[16,96] @ P[96,32]      (6 WMMAs, all operands packed)
//   scaled        = -J[e] * integ  -> f16 LDS (row-major, b128-coalescing)
//   t[16e,16r]    = scaled @ WB[32q,16r]      (1 WMMA)
//   res           = t - F ; loss += res^2/R
// ---------------------------------------------------------------------------
__global__ __launch_bounds__(32)
void vpinn_loss_kernel(const _Float16* __restrict__ Ap,
                       const _Float16* __restrict__ Pp,
                       const _Float16* __restrict__ WBf,
                       const float* __restrict__ J, const float* __restrict__ F,
                       float* __restrict__ out) {
    __shared__ _Float16 integ_h[16 * 32];  // [16 e][32 q] f16, -J-scaled
    __shared__ float    Jl[16];
    __shared__ float    red[32];

    const int lane = threadIdx.x;
    const int et   = blockIdx.x;
    const int e0   = et * 16;
    const int s    = blockIdx.y;

    if (lane < 16) {
        Jl[lane] = J[(size_t)(e0 + lane) * S_ + s];
        // get the F tile (960 B) flowing while the WMMAs run
        __builtin_prefetch(F + ((size_t)(e0 + lane) * S_ + s) * R_, 0, 0);
    }
    __syncthreads();

    // integrand[16e, 32q] = A[16,96] @ P[96,32]
    v8f acc0 = {}, acc1 = {};
    const _Float16* Abase = Ap + (size_t)(et * 3) * 512;
    const _Float16* Pbase = Pp + (size_t)(s * 6) * 512;
#pragma unroll
    for (int kt = 0; kt < 3; ++kt) {
        v16h a  = *(const v16h*)(Abase + (kt * 32 + lane) * 16);
        v16h b0 = *(const v16h*)(Pbase + ((kt * 2 + 0) * 32 + lane) * 16);
        v16h b1 = *(const v16h*)(Pbase + ((kt * 2 + 1) * 32 + lane) * 16);
        acc0 = __builtin_amdgcn_wmma_f32_16x16x32_f16(false, a, false, b0, (short)0, acc0, false, false);
        acc1 = __builtin_amdgcn_wmma_f32_16x16x32_f16(false, a, false, b1, (short)0, acc1, false, false);
    }

    // scatter C tiles with -J scaling, f16: lane = col q, vgpr j = row mb8+j
    const int nq  = lane & 15;
    const int mb8 = (lane >> 4) << 3;
#pragma unroll
    for (int j = 0; j < 8; ++j) {
        float jm = -Jl[mb8 + j];
        integ_h[(mb8 + j) * 32 + nq]      = (_Float16)(jm * acc0[j]);
        integ_h[(mb8 + j) * 32 + 16 + nq] = (_Float16)(jm * acc1[j]);
    }
    __syncthreads();

    // projection as one 16x16x32 WMMA: t = (-J*integ)[16,32] @ WB[32,16]
    const int khalf8 = (lane >> 4) << 3;
    v16h a2;
#pragma unroll
    for (int i = 0; i < 16; ++i) {
        int v = i >> 1;
        int k = ((v >> 2) << 4) + khalf8 + ((v & 3) << 1) + (i & 1);
        a2[i] = integ_h[(lane & 15) * 32 + k];
    }
    v16h bw = *(const v16h*)(WBf + lane * 16);
    v8f t = {};
    t = __builtin_amdgcn_wmma_f32_16x16x32_f16(false, a2, false, bw, (short)0, t, false, false);

    // residual: lane = col r, vgpr j = row mb8+j ; coalesced F reads
    float lacc = 0.0f;
    if (nq < R_) {
#pragma unroll
        for (int j = 0; j < 8; ++j) {
            int m = mb8 + j;
            float res = t[j] - F[((size_t)(e0 + m) * S_ + s) * R_ + nq];
            lacc = __builtin_fmaf(res, res, lacc);
        }
    }
    red[lane] = lacc;
    __syncthreads();
    if (lane == 0) {
        float ssum = 0.0f;
#pragma unroll
        for (int i = 0; i < 32; ++i) ssum += red[i];
        atomicAdd(out, ssum * (1.0f / R_));
    }
}

// ---------------------------------------------------------------------------
// Launch
// ---------------------------------------------------------------------------
extern "C" void kernel_launch(void* const* d_in, const int* in_sizes, int n_in,
                              void* d_out, int out_size, void* d_ws, size_t ws_size,
                              hipStream_t stream) {
    const float* nodes = (const float*)d_in[0];
    const float* W1    = (const float*)d_in[1];
    const float* b1    = (const float*)d_in[2];
    const float* W2    = (const float*)d_in[3];
    const float* b2    = (const float*)d_in[4];
    const float* W3    = (const float*)d_in[5];
    const float* b3    = (const float*)d_in[6];
    const float* W4    = (const float*)d_in[7];
    const float* b4    = (const float*)d_in[8];
    const float* Ixx   = (const float*)d_in[9];
    const float* Iyy   = (const float*)d_in[10];
    const float* BDD   = (const float*)d_in[11];
    const float* wq    = (const float*)d_in[12];
    const float* Base  = (const float*)d_in[13];
    const float* J     = (const float*)d_in[14];
    const float* F     = (const float*)d_in[15];
    float* out = (float*)d_out;

    float*    ev  = (float*)((char*)d_ws + WS_EV_OFF);
    _Float16* Wp  = (_Float16*)((char*)d_ws + WS_WP_OFF);
    _Float16* Pp  = (_Float16*)((char*)d_ws + WS_PP_OFF);
    _Float16* Ap  = (_Float16*)((char*)d_ws + WS_AP_OFF);
    _Float16* WBf = (_Float16*)((char*)d_ws + WS_WB_OFF);

    vpinn_zero_out<<<(out_size + 255) / 256, 256, 0, stream>>>(out, out_size);

    vpinn_pack_w<<<(2 * 32 * 32 * 16 + 255) / 256, 256, 0, stream>>>(W2, W3, Wp);
    vpinn_pack_p<<<(S_ * 3 * 2 * 32 * 16 + 255) / 256, 256, 0, stream>>>(Ixx, Iyy, Pp);
    vpinn_pack_wb<<<2, 256, 0, stream>>>(wq, Base, WBf);

    vpinn_mlp_kernel<<<dim3((E_ * NN_) / 16), 32, 0, stream>>>(
        nodes, W1, b1, Wp, b2, b3, W4, b4, ev);

    vpinn_pack_a<<<((E_ / 16) * 3 * 32 * 16 + 255) / 256, 256, 0, stream>>>(ev, BDD, Ap);

    vpinn_loss_kernel<<<dim3(E_ / 16, S_), 32, 0, stream>>>(
        Ap, Pp, WBf, J, F, out);
}
